// MultiHeadSelfAttention_6253472383351
// MI455X (gfx1250) — matrix operand order repaired and checked by reading, not compiled, not run on hardware
//
#include <hip/hip_runtime.h>
#include <hip/hip_bf16.h>

// MI455X / gfx1250 fused causal multi-head attention pipeline.
// All matmuls run on v_wmma_f32_16x16x32_bf16 (wave32 matrix core path).
// GEMMs stage weight tiles into LDS with GLOBAL_LOAD_ASYNC_TO_LDS_B128
// (ASYNCcnt path), shared by 4 waves per block.
// Workspace (bf16): Xbf | Wq | Wk | Wv | Wo | Q | K | Vt | Y = 48 MiB -> L2-resident.

#define BB 2
#define TT 2048
#define CC 1024
#define HH 16
#define DD 64
#define MM (BB * TT)   // 4096 rows

typedef __attribute__((ext_vector_type(16))) __bf16 v16bf;
typedef __attribute__((ext_vector_type(8)))  __bf16 v8bf;
typedef __attribute__((ext_vector_type(8)))  float  v8f;

// ---------------------------------------------------------------- WMMA helper
__device__ __forceinline__ v8f wmma_bf16(v16bf a, v16bf b, v8f c) {
  // (neg_a, A, neg_b, B, c_mod, C, reuse_a, reuse_b)
  return __builtin_amdgcn_wmma_f32_16x16x32_bf16(false, a, false, b,
                                                 (short)0, c, false, false);
}

// A-matrix 16x32 bf16 fragment (contiguous k order, same as the B loads): lane m=lane&15, half=lane>>4.
// frag[0..7] = row[half*16 + j], frag[8..15] = row[half*16 + 8 + j].
__device__ __forceinline__ v16bf load_a_frag(const __bf16* row, int hi) {
  v8bf lo = *(const v8bf*)(row + hi * 16);
  v8bf hh = *(const v8bf*)(row + hi * 16 + 8);
  v16bf r;
#pragma unroll
  for (int i = 0; i < 8; ++i) { r[i] = lo[i]; r[i + 8] = hh[i]; }
  return r;
}

// Async copy 16B from global to LDS (ASYNCcnt path, GV addressing mode).
__device__ __forceinline__ void async_g2l_b128(unsigned lds_byte_addr,
                                               const void* gaddr) {
  asm volatile("global_load_async_to_lds_b128 %0, %1, off"
               :: "v"(lds_byte_addr),
                  "v"((unsigned long long)(size_t)gaddr)
               : "memory");
}
__device__ __forceinline__ void wait_async0() {
  asm volatile("s_wait_asynccnt 0" ::: "memory");
}

__device__ __forceinline__ float half_max(float v) {
  v = fmaxf(v, __shfl_xor(v, 1, 32));
  v = fmaxf(v, __shfl_xor(v, 2, 32));
  v = fmaxf(v, __shfl_xor(v, 4, 32));
  v = fmaxf(v, __shfl_xor(v, 8, 32));
  return v;
}
__device__ __forceinline__ float half_sum(float v) {
  v += __shfl_xor(v, 1, 32);
  v += __shfl_xor(v, 2, 32);
  v += __shfl_xor(v, 4, 32);
  v += __shfl_xor(v, 8, 32);
  return v;
}

// ------------------------------------------------------------ fp32 -> bf16
__global__ void cvt_f32_to_bf16(const float* __restrict__ src,
                                __bf16* __restrict__ dst, int n) {
  for (int i = blockIdx.x * blockDim.x + threadIdx.x; i < n;
       i += gridDim.x * blockDim.x)
    dst[i] = (__bf16)src[i];
}

// V stored transposed per head: Vt[b][h][d][t]  (row-contiguous in t)
__device__ __forceinline__ void store_vt(__bf16* Vt, int row, int col, float v) {
  int b = row >> 11;          // / TT
  int t = row & (TT - 1);
  int h = col >> 6;           // / DD
  int d = col & (DD - 1);
  Vt[(((size_t)b * HH + h) * DD + d) * (size_t)TT + t] = (__bf16)v;
}

// LDS weight tile: 32 rows (output cols) x 32 halfs per k-step.
// Row stride 48 halfs = 96 B: 32B-aligned v16bf reads, 24-bank stride.
#define WT_STRIDE 48

// ------------------------------------------------- QKV projection (bf16 WMMA)
// Block = 4 waves -> 128x32 output tile. Weight k-slab staged in LDS via
// async loads, shared by all 4 waves; activations streamed from L2.
__global__ __launch_bounds__(128) void gemm_qkv_kernel(
    const __bf16* __restrict__ X,
    const __bf16* __restrict__ Wq, const __bf16* __restrict__ Wk,
    const __bf16* __restrict__ Wv,
    const float* __restrict__ bq, const float* __restrict__ bk,
    const float* __restrict__ bv,
    __bf16* __restrict__ Qb, __bf16* __restrict__ Kb, __bf16* __restrict__ Vt) {
  __shared__ __align__(32) __bf16 wtile[32 * WT_STRIDE];
  const int tid = threadIdx.x;
  const int wid = tid >> 5;
  const int lane = tid & 31;
  const int l15 = lane & 15;
  const int hi = lane >> 4;
  const int m0 = blockIdx.x * 128 + wid * 32;
  const int n0 = blockIdx.y * 32;
  const int which = blockIdx.z;
  const __bf16* W   = (which == 0) ? Wq : (which == 1 ? Wk : Wv);
  const float* bias = (which == 0) ? bq : (which == 1 ? bk : bv);

  // staging assignment: thread i copies 16B chunk (i&3) of weight row (i>>2)
  const int srow = tid >> 2;
  const int schunk = tid & 3;
  const __bf16* wsrc = W + (size_t)(n0 + srow) * CC + schunk * 8;
  const unsigned ldsdst =
      (unsigned)(size_t)wtile + (unsigned)(srow * (WT_STRIDE * 2) + schunk * 16);

  v8f a00 = {}, a01 = {}, a10 = {}, a11 = {};
  const __bf16* x0 = X + (size_t)(m0 + l15) * CC;
  const __bf16* x1 = x0 + (size_t)16 * CC;

  for (int k0 = 0; k0 < CC; k0 += 32) {
    async_g2l_b128(ldsdst, wsrc + k0);            // stage 32x32 weight slab
    __builtin_prefetch(x0 + k0 + 128, 0, 1);      // global_prefetch on A stream
    __builtin_prefetch(x1 + k0 + 128, 0, 1);
    wait_async0();
    __syncthreads();

    v16bf A0 = load_a_frag(x0 + k0, hi);
    v16bf A1 = load_a_frag(x1 + k0, hi);
    v16bf B0 = *(const v16bf*)(wtile + l15 * WT_STRIDE + hi * 16);
    v16bf B1 = *(const v16bf*)(wtile + (16 + l15) * WT_STRIDE + hi * 16);
    a00 = wmma_bf16(A0, B0, a00);
    a01 = wmma_bf16(A0, B1, a01);
    a10 = wmma_bf16(A1, B0, a10);
    a11 = wmma_bf16(A1, B1, a11);
    __syncthreads();                              // before next slab overwrite
  }

  const int col0 = n0 + l15;
  const int col1 = col0 + 16;
  const float bv0 = bias[col0], bv1 = bias[col1];
#pragma unroll
  for (int r = 0; r < 8; ++r) {
    int row0 = m0 + r + 8 * hi;      // C/D layout: VGPR r, half -> row r+8*half
    int row1 = row0 + 16;
    float v00 = a00[r] + bv0, v01 = a01[r] + bv1;
    float v10 = a10[r] + bv0, v11 = a11[r] + bv1;
    if (which == 2) {
      store_vt(Vt, row0, col0, v00);
      store_vt(Vt, row0, col1, v01);
      store_vt(Vt, row1, col0, v10);
      store_vt(Vt, row1, col1, v11);
    } else {
      __bf16* O = (which == 0) ? Qb : Kb;
      O[(size_t)row0 * CC + col0] = (__bf16)v00;
      O[(size_t)row0 * CC + col1] = (__bf16)v01;
      O[(size_t)row1 * CC + col0] = (__bf16)v10;
      O[(size_t)row1 * CC + col1] = (__bf16)v11;
    }
  }
}

// ------------------------------------------------------- flash attention core
// One wave handles a 16-row Q tile for one (b,h); iterates causal k-tiles of 32.
__global__ __launch_bounds__(32) void attn_kernel(
    const __bf16* __restrict__ Qb, const __bf16* __restrict__ Kb,
    const __bf16* __restrict__ Vt, __bf16* __restrict__ Yb) {
  __shared__ __align__(32) __bf16 pbuf[16 * 32];  // P tile C-layout -> A-layout
  const int q0 = blockIdx.x * 16;
  const int h = blockIdx.y;
  const int b = blockIdx.z;
  const int lane = threadIdx.x & 31;
  const int l15 = lane & 15;
  const int hi = lane >> 4;

  const __bf16* qrow = Qb + (size_t)(b * TT + q0 + l15) * CC + h * DD;
  v16bf qa0 = load_a_frag(qrow, hi);        // d = 0..31
  v16bf qa1 = load_a_frag(qrow + 32, hi);   // d = 32..63

  float mrow[8], lrow[8];
#pragma unroll
  for (int r = 0; r < 8; ++r) { mrow[r] = -3.0e38f; lrow[r] = 0.f; }
  v8f o0 = {}, o1 = {}, o2 = {}, o3 = {};
  const float scale = 0.125f;  // 1/sqrt(64)

  for (int t0 = 0; t0 <= q0; t0 += 32) {
    // S = Q K^T : B columns (n=t) are rows of K -> contiguous loads
    const __bf16* kr0 = Kb + (size_t)(b * TT + t0 + l15) * CC + h * DD + hi * 16;
    const __bf16* kr1 = kr0 + (size_t)16 * CC;
    v16bf kb00 = *(const v16bf*)(kr0);        // t tile 0, d 0..31
    v16bf kb01 = *(const v16bf*)(kr0 + 32);   // t tile 0, d 32..63
    v16bf kb10 = *(const v16bf*)(kr1);
    v16bf kb11 = *(const v16bf*)(kr1 + 32);
    v8f s0 = {}, s1 = {};
    s0 = wmma_bf16(qa0, kb00, s0);
    s0 = wmma_bf16(qa1, kb01, s0);
    s1 = wmma_bf16(qa0, kb10, s1);
    s1 = wmma_bf16(qa1, kb11, s1);

    // online softmax over 32 columns (half-wave width-16 reductions)
#pragma unroll
    for (int r = 0; r < 8; ++r) {
      int row = q0 + r + 8 * hi;
      float v0 = s0[r] * scale;
      float v1 = s1[r] * scale;
      if (t0 + l15 > row)      v0 = -3.0e38f;   // causal mask
      if (t0 + 16 + l15 > row) v1 = -3.0e38f;
      float mx = half_max(fmaxf(v0, v1));
      float mnew = fmaxf(mrow[r], mx);
      float alpha = __expf(mrow[r] - mnew);
      float p0 = __expf(v0 - mnew);
      float p1 = __expf(v1 - mnew);
      float ps = half_sum(p0 + p1);
      lrow[r] = lrow[r] * alpha + ps;
      mrow[r] = mnew;
      o0[r] *= alpha; o1[r] *= alpha; o2[r] *= alpha; o3[r] *= alpha;
      pbuf[(r + 8 * hi) * 32 + l15]      = (__bf16)p0;
      pbuf[(r + 8 * hi) * 32 + 16 + l15] = (__bf16)p1;
    }
    __syncthreads();

    // reload P in A-fragment layout from LDS
    v16bf pa;
    {
      const __bf16* prow = &pbuf[l15 * 32];
      v8bf lo = *(const v8bf*)(prow + hi * 16);
      v8bf hh = *(const v8bf*)(prow + hi * 16 + 8);
#pragma unroll
      for (int i = 0; i < 8; ++i) { pa[i] = lo[i]; pa[8 + i] = hh[i]; }
    }

    // O += P * V : B columns (n=d) are rows of Vt -> contiguous loads
    const __bf16* vbase =
        Vt + (size_t)(b * HH + h) * DD * (size_t)TT + t0 + hi * 16;
    o0 = wmma_bf16(pa, *(const v16bf*)(vbase + (size_t)(0 * 16 + l15) * TT), o0);
    o1 = wmma_bf16(pa, *(const v16bf*)(vbase + (size_t)(1 * 16 + l15) * TT), o1);
    o2 = wmma_bf16(pa, *(const v16bf*)(vbase + (size_t)(2 * 16 + l15) * TT), o2);
    o3 = wmma_bf16(pa, *(const v16bf*)(vbase + (size_t)(3 * 16 + l15) * TT), o3);
    __syncthreads();
  }

#pragma unroll
  for (int r = 0; r < 8; ++r) {
    float inv = 1.0f / lrow[r];
    int rowg = q0 + r + 8 * hi;
    size_t base = (size_t)(b * TT + rowg) * CC + h * DD + l15;
    Yb[base + 0]  = (__bf16)(o0[r] * inv);
    Yb[base + 16] = (__bf16)(o1[r] * inv);
    Yb[base + 32] = (__bf16)(o2[r] * inv);
    Yb[base + 48] = (__bf16)(o3[r] * inv);
  }
}

// ------------------------------------------------------- output projection
__global__ __launch_bounds__(128) void gemm_out_kernel(
    const __bf16* __restrict__ Yb, const __bf16* __restrict__ Wo,
    const float* __restrict__ bo, float* __restrict__ out) {
  __shared__ __align__(32) __bf16 wtile[32 * WT_STRIDE];
  const int tid = threadIdx.x;
  const int wid = tid >> 5;
  const int lane = tid & 31;
  const int l15 = lane & 15;
  const int hi = lane >> 4;
  const int m0 = blockIdx.x * 128 + wid * 32;
  const int n0 = blockIdx.y * 32;

  const int srow = tid >> 2;
  const int schunk = tid & 3;
  const __bf16* wsrc = Wo + (size_t)(n0 + srow) * CC + schunk * 8;
  const unsigned ldsdst =
      (unsigned)(size_t)wtile + (unsigned)(srow * (WT_STRIDE * 2) + schunk * 16);

  v8f a00 = {}, a01 = {}, a10 = {}, a11 = {};
  const __bf16* x0 = Yb + (size_t)(m0 + l15) * CC;
  const __bf16* x1 = x0 + (size_t)16 * CC;

  for (int k0 = 0; k0 < CC; k0 += 32) {
    async_g2l_b128(ldsdst, wsrc + k0);
    __builtin_prefetch(x0 + k0 + 128, 0, 1);
    __builtin_prefetch(x1 + k0 + 128, 0, 1);
    wait_async0();
    __syncthreads();

    v16bf A0 = load_a_frag(x0 + k0, hi);
    v16bf A1 = load_a_frag(x1 + k0, hi);
    v16bf B0 = *(const v16bf*)(wtile + l15 * WT_STRIDE + hi * 16);
    v16bf B1 = *(const v16bf*)(wtile + (16 + l15) * WT_STRIDE + hi * 16);
    a00 = wmma_bf16(A0, B0, a00);
    a01 = wmma_bf16(A0, B1, a01);
    a10 = wmma_bf16(A1, B0, a10);
    a11 = wmma_bf16(A1, B1, a11);
    __syncthreads();
  }

  const float b0v = bo[n0 + l15], b1v = bo[n0 + 16 + l15];
#pragma unroll
  for (int r = 0; r < 8; ++r) {
    int row0 = m0 + r + 8 * hi, row1 = row0 + 16;
    out[(size_t)row0 * CC + n0 + l15]      = a00[r] + b0v;
    out[(size_t)row0 * CC + n0 + 16 + l15] = a01[r] + b1v;
    out[(size_t)row1 * CC + n0 + l15]      = a10[r] + b0v;
    out[(size_t)row1 * CC + n0 + 16 + l15] = a11[r] + b1v;
  }
}

// ---------------------------------------------------------------- launcher
extern "C" void kernel_launch(void* const* d_in, const int* in_sizes, int n_in,
                              void* d_out, int out_size, void* d_ws,
                              size_t ws_size, hipStream_t stream) {
  (void)in_sizes; (void)n_in; (void)out_size; (void)ws_size;
  const float* x  = (const float*)d_in[0];
  const float* Wq = (const float*)d_in[1];
  const float* bq = (const float*)d_in[2];
  const float* Wk = (const float*)d_in[3];
  const float* bk = (const float*)d_in[4];
  const float* Wv = (const float*)d_in[5];
  const float* bv = (const float*)d_in[6];
  const float* Wo = (const float*)d_in[7];
  const float* bo = (const float*)d_in[8];
  float* out = (float*)d_out;

  char* p = (char*)d_ws;
  const size_t SZ_ACT = (size_t)MM * CC * sizeof(__bf16);  // 8 MiB
  const size_t SZ_W   = (size_t)CC * CC * sizeof(__bf16);  // 2 MiB
  __bf16* Xbf = (__bf16*)p; p += SZ_ACT;
  __bf16* Wqb = (__bf16*)p; p += SZ_W;
  __bf16* Wkb = (__bf16*)p; p += SZ_W;
  __bf16* Wvb = (__bf16*)p; p += SZ_W;
  __bf16* Wob = (__bf16*)p; p += SZ_W;
  __bf16* Qb  = (__bf16*)p; p += SZ_ACT;
  __bf16* Kb  = (__bf16*)p; p += SZ_ACT;
  __bf16* Vtb = (__bf16*)p; p += SZ_ACT;
  __bf16* Yb  = (__bf16*)p; p += SZ_ACT;   // total 48 MiB

  cvt_f32_to_bf16<<<dim3(2048), 256, 0, stream>>>(x,  Xbf, MM * CC);
  cvt_f32_to_bf16<<<dim3(1024), 256, 0, stream>>>(Wq, Wqb, CC * CC);
  cvt_f32_to_bf16<<<dim3(1024), 256, 0, stream>>>(Wk, Wkb, CC * CC);
  cvt_f32_to_bf16<<<dim3(1024), 256, 0, stream>>>(Wv, Wvb, CC * CC);
  cvt_f32_to_bf16<<<dim3(1024), 256, 0, stream>>>(Wo, Wob, CC * CC);

  gemm_qkv_kernel<<<dim3(MM / 128, CC / 32, 3), 128, 0, stream>>>(
      Xbf, Wqb, Wkb, Wvb, bq, bk, bv, Qb, Kb, Vtb);

  attn_kernel<<<dim3(TT / 16, HH, BB), 32, 0, stream>>>(Qb, Kb, Vtb, Yb);

  gemm_out_kernel<<<dim3(MM / 128, CC / 32), 128, 0, stream>>>(Yb, Wob, bo, out);
}
